// FocalAttention_11622181503354
// MI455X (gfx1250) — compile-verified, matrix operand order checked by reading, not run on hardware
//
#include <hip/hip_runtime.h>
#include <hip/hip_bf16.h>

typedef float  v8f   __attribute__((ext_vector_type(8)));
typedef __bf16 v16bf __attribute__((ext_vector_type(16)));
typedef __bf16 v4bf  __attribute__((ext_vector_type(4)));
typedef __bf16 v2bf  __attribute__((ext_vector_type(2)));
typedef int    v4i_  __attribute__((ext_vector_type(4)));

#define TILE_M 128
#define TILE_N 128
#define TILE_K 32

#if __has_builtin(__builtin_amdgcn_global_load_async_to_lds_b128)
#define HAVE_ASYNC_LDS 1
#endif

#if defined(HAVE_ASYNC_LDS)
// builtin expects int4* in addrspace(1)/(3); HIP prints as1 as `__device__`.
#define GLOBAL_AS(p) ((__attribute__((address_space(1))) v4i_*)(p))
#define LDS_AS(p)    ((__attribute__((address_space(3))) v4i_*)(p))
#if __has_builtin(__builtin_amdgcn_s_wait_asynccnt)
#define WAIT_ASYNC(n) __builtin_amdgcn_s_wait_asynccnt(n)
#else
#define WAIT_ASYNC(n) asm volatile("s_wait_asynccnt %0" :: "i"(n) : "memory")
#endif
#endif

// ---------------------------------------------------------------------------
// Elementwise fp32 -> split bf16 (hi + residual lo), done once per tensor so
// the GEMM hot loop has zero conversion VALU and tiles are raw async-copyable.
// ---------------------------------------------------------------------------
__global__ __launch_bounds__(256)
void f32_to_bf16_split(const float* __restrict__ in, __bf16* __restrict__ hi,
                       __bf16* __restrict__ lo, int n4) {
  int i = blockIdx.x * blockDim.x + threadIdx.x;
  if (i >= n4) return;
  float4 f = ((const float4*)in)[i];
  v4bf h, l;
  h.x = (__bf16)f.x;  l.x = (__bf16)(f.x - (float)h.x);
  h.y = (__bf16)f.y;  l.y = (__bf16)(f.y - (float)h.y);
  h.z = (__bf16)f.z;  l.z = (__bf16)(f.z - (float)h.z);
  h.w = (__bf16)f.w;  l.w = (__bf16)(f.w - (float)h.w);
  ((v4bf*)hi)[i] = h;
  ((v4bf*)lo)[i] = l;
}

// ---------------------------------------------------------------------------
// C[M,N] = (Ahi+Alo)[M,K] @ (Bhi+Blo)[N,K]^T + bias[N]   (bf16 split inputs)
// Double-buffered LDS; tiles streamed with GLOBAL_LOAD_ASYNC_TO_LDS_B128
// (ASYNCcnt) when available. 3 WMMAs (hh + hl + lh) per 16x16 tile ~ fp32.
// ---------------------------------------------------------------------------
__global__ __launch_bounds__(256)
void gemm_bf16x3(const __bf16* __restrict__ Ahi, const __bf16* __restrict__ Alo,
                 const __bf16* __restrict__ Bhi, const __bf16* __restrict__ Blo,
                 const float* __restrict__ bias, float* __restrict__ C,
                 int M, int N, int K) {
  __shared__ __align__(32) __bf16 sAhi[2][TILE_M * TILE_K];
  __shared__ __align__(32) __bf16 sAlo[2][TILE_M * TILE_K];
  __shared__ __align__(32) __bf16 sBhi[2][TILE_N * TILE_K];
  __shared__ __align__(32) __bf16 sBlo[2][TILE_N * TILE_K];

  const int tid   = threadIdx.x;
  const int lane  = tid & 31;
  const int wave  = tid >> 5;     // 0..7
  const int waveM = wave >> 2;    // 0..1 -> 64-row slab
  const int waveN = wave & 3;     // 0..3 -> 32-col slab
  const int m0 = blockIdx.y * TILE_M;
  const int n0 = blockIdx.x * TILE_N;

  // cooperative copy mapping: thread owns 32 contiguous bytes of one tile row
  const int crow = tid >> 1;          // 0..127
  const int ccol = (tid & 1) * 16;    // element offset (bf16) within 32-wide row

  // WMMA fragment addressing (contiguous k order, same for A and B):
  // lane l -> row/col (l&15), K-base (l>>4)*16, 16 contiguous bf16.
  const int fr = lane & 15;
  const int kb = (lane >> 4) * 16;

  // copy one K-step (4 sub-tiles of 128x32 bf16 = 32KB) into LDS buffer `buf`
  auto issue_tile = [&](int kk, int buf) {
    const __bf16* gA0 = Ahi + (size_t)(m0 + crow) * K + kk + ccol;
    const __bf16* gA1 = Alo + (size_t)(m0 + crow) * K + kk + ccol;
    const __bf16* gB0 = Bhi + (size_t)(n0 + crow) * K + kk + ccol;
    const __bf16* gB1 = Blo + (size_t)(n0 + crow) * K + kk + ccol;
    __bf16* lA0 = &sAhi[buf][crow * TILE_K + ccol];
    __bf16* lA1 = &sAlo[buf][crow * TILE_K + ccol];
    __bf16* lB0 = &sBhi[buf][crow * TILE_K + ccol];
    __bf16* lB1 = &sBlo[buf][crow * TILE_K + ccol];
#if defined(HAVE_ASYNC_LDS)
    // 8 wave-wide async ops per wave per K-step; in-order completion.
    __builtin_amdgcn_global_load_async_to_lds_b128(GLOBAL_AS(gA0), LDS_AS(lA0), 0, 0);
    __builtin_amdgcn_global_load_async_to_lds_b128(GLOBAL_AS(gA0), LDS_AS(lA0), 16, 0);
    __builtin_amdgcn_global_load_async_to_lds_b128(GLOBAL_AS(gA1), LDS_AS(lA1), 0, 0);
    __builtin_amdgcn_global_load_async_to_lds_b128(GLOBAL_AS(gA1), LDS_AS(lA1), 16, 0);
    __builtin_amdgcn_global_load_async_to_lds_b128(GLOBAL_AS(gB0), LDS_AS(lB0), 0, 0);
    __builtin_amdgcn_global_load_async_to_lds_b128(GLOBAL_AS(gB0), LDS_AS(lB0), 16, 0);
    __builtin_amdgcn_global_load_async_to_lds_b128(GLOBAL_AS(gB1), LDS_AS(lB1), 0, 0);
    __builtin_amdgcn_global_load_async_to_lds_b128(GLOBAL_AS(gB1), LDS_AS(lB1), 16, 0);
#else
    ((uint4*)lA0)[0] = ((const uint4*)gA0)[0];
    ((uint4*)lA0)[1] = ((const uint4*)gA0)[1];
    ((uint4*)lA1)[0] = ((const uint4*)gA1)[0];
    ((uint4*)lA1)[1] = ((const uint4*)gA1)[1];
    ((uint4*)lB0)[0] = ((const uint4*)gB0)[0];
    ((uint4*)lB0)[1] = ((const uint4*)gB0)[1];
    ((uint4*)lB1)[0] = ((const uint4*)gB1)[0];
    ((uint4*)lB1)[1] = ((const uint4*)gB1)[1];
#endif
  };

  v8f acc[4][2] = {};

  const int nk = K / TILE_K;
  issue_tile(0, 0);

  for (int ik = 0; ik < nk; ++ik) {
    const int cur = ik & 1;
    if (ik + 1 < nk) {
      issue_tile((ik + 1) * TILE_K, cur ^ 1);   // prefetch behind the WMMAs
#if defined(HAVE_ASYNC_LDS)
      WAIT_ASYNC(8);                            // current tile's 8 ops done
#endif
    } else {
#if defined(HAVE_ASYNC_LDS)
      WAIT_ASYNC(0);
#endif
    }
    __syncthreads();   // tile data visible to all waves

    v16bf aH[4], aL[4], bH[2], bL[2];
    #pragma unroll
    for (int i = 0; i < 4; ++i) {
      int r = waveM * 64 + i * 16 + fr;
      aH[i] = *(const v16bf*)&sAhi[cur][r * TILE_K + kb];
      aL[i] = *(const v16bf*)&sAlo[cur][r * TILE_K + kb];
    }
    #pragma unroll
    for (int j = 0; j < 2; ++j) {
      int c = waveN * 32 + j * 16 + fr;
      bH[j] = *(const v16bf*)&sBhi[cur][c * TILE_K + kb];
      bL[j] = *(const v16bf*)&sBlo[cur][c * TILE_K + kb];
    }

    #pragma unroll
    for (int i = 0; i < 4; ++i) {
      #pragma unroll
      for (int j = 0; j < 2; ++j) {
        acc[i][j] = __builtin_amdgcn_wmma_f32_16x16x32_bf16(
            false, aH[i], false, bH[j], (short)0, acc[i][j], false, false);
        acc[i][j] = __builtin_amdgcn_wmma_f32_16x16x32_bf16(
            false, aH[i], false, bL[j], (short)0, acc[i][j], false, false);
        acc[i][j] = __builtin_amdgcn_wmma_f32_16x16x32_bf16(
            false, aL[i], false, bH[j], (short)0, acc[i][j], false, false);
      }
    }
    __syncthreads();   // reads done before this buffer is overwritten (ik+2)
  }

  // epilogue: bias add + store (C layout: VGPR r -> M=r / r+8 by lane half)
  const int rowAdd = (lane >> 4) * 8;
  #pragma unroll
  for (int i = 0; i < 4; ++i) {
    #pragma unroll
    for (int j = 0; j < 2; ++j) {
      int gcol = n0 + waveN * 32 + j * 16 + fr;
      float bv = bias[gcol];
      #pragma unroll
      for (int r = 0; r < 8; ++r) {
        int grow = m0 + waveM * 64 + i * 16 + rowAdd + r;
        C[(size_t)grow * N + gcol] = acc[i][j][r] + bv;
      }
    }
  }
}

#define T_SEQ  2048
#define DMODEL 1024
#define NHEADS 16
#define HDIM   64
#define RAD    4
#define NWIN   (2 * RAD + 1)

// One wave per (b,h,t): 32 lanes x 2 dims cover hd=64; <=9 keys per query.
// Emits split-bf16 output directly for the projection GEMM.
__global__ __launch_bounds__(256)
void focal_attn(const float* __restrict__ qkv, __bf16* __restrict__ out_hi,
                __bf16* __restrict__ out_lo) {
  const int lane = threadIdx.x & 31;
  const int gw = blockIdx.x * 8 + (threadIdx.x >> 5);
  const int t  = gw % T_SEQ;
  const int bh = gw / T_SEQ;
  const int h  = bh % NHEADS;
  const int b  = bh / NHEADS;

  const size_t rowBase = ((size_t)(b * T_SEQ + t)) * (3 * DMODEL);
  const int    hoff    = h * HDIM + lane * 2;
  const float2 q2 = *(const float2*)(qkv + rowBase + hoff);
  const float  scale = 0.125f; // 1/sqrt(64)

  float  s[NWIN];
  float2 vv[NWIN];
  #pragma unroll
  for (int j = 0; j < NWIN; ++j) {
    int  tk = t - RAD + j;
    bool ok = (tk >= 0) && (tk < T_SEQ);
    int  tc = ok ? tk : t;
    const size_t kbase = ((size_t)(b * T_SEQ + tc)) * (3 * DMODEL) + hoff;
    float2 k2 = *(const float2*)(qkv + kbase + DMODEL);
    vv[j]     = *(const float2*)(qkv + kbase + 2 * DMODEL);
    float p = q2.x * k2.x + q2.y * k2.y;
    #pragma unroll
    for (int off = 16; off > 0; off >>= 1) p += __shfl_xor(p, off, 32);
    s[j] = ok ? p * scale : -__builtin_inff();
  }

  float m = s[0];
  #pragma unroll
  for (int j = 1; j < NWIN; ++j) m = fmaxf(m, s[j]);
  float sum = 0.0f;
  #pragma unroll
  for (int j = 0; j < NWIN; ++j) { float e = __expf(s[j] - m); s[j] = e; sum += e; }
  float inv = 1.0f / sum;

  float2 o = make_float2(0.0f, 0.0f);
  #pragma unroll
  for (int j = 0; j < NWIN; ++j) { o.x += s[j] * vv[j].x; o.y += s[j] * vv[j].y; }
  o.x *= inv; o.y *= inv;

  v2bf hi2, lo2;
  hi2.x = (__bf16)o.x;  lo2.x = (__bf16)(o.x - (float)hi2.x);
  hi2.y = (__bf16)o.y;  lo2.y = (__bf16)(o.y - (float)hi2.y);
  const size_t oidx = ((size_t)(b * T_SEQ + t)) * DMODEL + hoff;
  *(v2bf*)(out_hi + oidx) = hi2;
  *(v2bf*)(out_lo + oidx) = lo2;
}

extern "C" void kernel_launch(void* const* d_in, const int* in_sizes, int n_in,
                              void* d_out, int out_size, void* d_ws, size_t ws_size,
                              hipStream_t stream) {
  const float* x      = (const float*)d_in[0];
  const float* w_qkv  = (const float*)d_in[1];
  const float* b_qkv  = (const float*)d_in[2];
  const float* w_proj = (const float*)d_in[3];
  const float* b_proj = (const float*)d_in[4];
  float* out = (float*)d_out;

  const int B = 4, T = T_SEQ, D = DMODEL;
  const int M = B * T;                         // 8192 rows
  const size_t nX  = (size_t)M * D;            // 8,388,608
  const size_t nWq = (size_t)3 * D * D;        // 3,145,728
  const size_t nWp = (size_t)D * D;            // 1,048,576

  // workspace layout (all offsets naturally aligned)
  char* w = (char*)d_ws;
  float*  qkv     = (float*)w;    w += (size_t)M * 3 * D * sizeof(float);  // 96 MB
  __bf16* x_hi    = (__bf16*)w;   w += nX  * sizeof(__bf16);
  __bf16* x_lo    = (__bf16*)w;   w += nX  * sizeof(__bf16);
  __bf16* wq_hi   = (__bf16*)w;   w += nWq * sizeof(__bf16);
  __bf16* wq_lo   = (__bf16*)w;   w += nWq * sizeof(__bf16);
  __bf16* wp_hi   = (__bf16*)w;   w += nWp * sizeof(__bf16);
  __bf16* wp_lo   = (__bf16*)w;   w += nWp * sizeof(__bf16);
  __bf16* attn_hi = (__bf16*)w;   w += nX  * sizeof(__bf16);
  __bf16* attn_lo = (__bf16*)w;

  dim3 blk(256);

  // one-shot split-bf16 conversion of activations and weights
  f32_to_bf16_split<<<dim3((unsigned)(nX  / 4 / 256)), blk, 0, stream>>>(x,      x_hi, x_lo, (int)(nX  / 4));
  f32_to_bf16_split<<<dim3((unsigned)(nWq / 4 / 256)), blk, 0, stream>>>(w_qkv,  wq_hi, wq_lo, (int)(nWq / 4));
  f32_to_bf16_split<<<dim3((unsigned)(nWp / 4 / 256)), blk, 0, stream>>>(w_proj, wp_hi, wp_lo, (int)(nWp / 4));

  // qkv = x @ w_qkv^T + b_qkv
  gemm_bf16x3<<<dim3((3 * D) / TILE_N, M / TILE_M), blk, 0, stream>>>(
      x_hi, x_lo, wq_hi, wq_lo, b_qkv, qkv, M, 3 * D, D);

  // local-window attention (radius 4), emits split-bf16
  focal_attn<<<dim3((B * NHEADS * T) / 8), blk, 0, stream>>>(qkv, attn_hi, attn_lo);

  // out = attn @ w_proj^T + b_proj
  gemm_bf16x3<<<dim3(D / TILE_N, M / TILE_M), blk, 0, stream>>>(
      attn_hi, attn_lo, wp_hi, wp_lo, b_proj, out, M, D, D);
}
